// MapEncoder_71949292142596
// MI455X (gfx1250) — compile-verified, matrix operand order checked by reading, not run on hardware
//
#include <hip/hip_runtime.h>
#include <hip/hip_bf16.h>

#define NN   200000
#define CC   128
#define SSC  6
#define EPP  100000
#define ELL  25000
#define KMETA 160
#define EPSV 1e-5f

typedef __attribute__((ext_vector_type(16))) __bf16 v16bf;
typedef __attribute__((ext_vector_type(4)))  __bf16 v4bf;
typedef __attribute__((ext_vector_type(8)))  float  v8f;
typedef __attribute__((ext_vector_type(4)))  float  v4f;
typedef __attribute__((ext_vector_type(4)))  unsigned int v4u;

union Frag { v4u q[2]; v16bf v; };

__device__ __forceinline__ v16bf load_frag16(const __bf16* p) {
    // 8 contiguous bf16 (lane's K+0..7 half) + 8 contiguous bf16 (K+16..23)
    Frag f;
    f.q[0] = *reinterpret_cast<const v4u*>(p);
    f.q[1] = *reinterpret_cast<const v4u*>(p + 16);
    return f.v;
}

// Async 16B global -> LDS copy (per-lane addresses), tracked by ASYNCcnt.
__device__ __forceinline__ void async_g2l_b128(const __bf16* gaddr, const __bf16* laddr) {
    unsigned lds = (unsigned)(uintptr_t)laddr;   // low 32 bits = wave-relative LDS byte addr
    asm volatile("global_load_async_to_lds_b128 %0, %1, off"
                 :: "v"(lds), "v"(gaddr) : "memory");
}
__device__ __forceinline__ void wait_asynccnt0() {
    asm volatile("s_wait_asynccnt 0x0" ::: "memory");
}

// Native no-return f32 atomic add (STOREcnt-tracked; drained at s_endpgm).
__device__ __forceinline__ void atomic_add_f32g(float* addr, float v) {
    asm volatile("global_atomic_add_f32 %0, %1, off" :: "v"(addr), "v"(v) : "memory");
}

__device__ __forceinline__ float wave_sum32(float v) {
    #pragma unroll
    for (int off = 16; off > 0; off >>= 1) v += __shfl_xor(v, off, 32);
    return v;
}

// ---------------- weight prep: f32 [M][K][N] -> bf16 [M][N][K] (transposed) -------------
__global__ __launch_bounds__(256) void k_transpose_many(const float* __restrict__ src,
                                                        __bf16* __restrict__ dst, int M) {
    int idx = blockIdx.x * 256 + threadIdx.x;
    int total = M * CC * CC;
    if (idx >= total) return;
    int m   = idx / (CC * CC);
    int rem = idx - m * (CC * CC);
    int n   = rem / CC;
    int k   = rem - n * CC;
    dst[idx] = (__bf16)src[(size_t)m * CC * CC + (size_t)k * CC + n];
}

// w_meta [132][128] -> bf16 [128][160] transposed, zero-padded K 132..159
__global__ __launch_bounds__(256) void k_transpose_meta(const float* __restrict__ src,
                                                        __bf16* __restrict__ dst) {
    int idx = blockIdx.x * 256 + threadIdx.x;
    if (idx >= CC * KMETA) return;
    int n = idx / KMETA;
    int k = idx - n * KMETA;
    float v = (k < CC + 4) ? src[(size_t)k * CC + n] : 0.0f;
    dst[idx] = (__bf16)v;
}

// ---------------- input branch hidden: h = relu(x @ W[2,C] + b) -> bf16 ----------------
__global__ __launch_bounds__(256) void k_branch(const float* __restrict__ x,
                                                const float* __restrict__ w,
                                                const float* __restrict__ bias,
                                                __bf16* __restrict__ out) {
    int idx = blockIdx.x * 256 + threadIdx.x;
    if (idx >= NN * CC) return;
    int row = idx / CC;
    int c   = idx - row * CC;
    float h = fmaf(x[row * 2 + 1], w[CC + c], fmaf(x[row * 2], w[c], bias[c]));
    out[idx] = (__bf16)fmaxf(h, 0.0f);
}

// ---------------- dense GEMM: Y[N,128] = Xbf16[N,K] @ Wt_bf16[128,K] -------------------
// grid.x = rows/16, block = 256 (8 waves = 8 column tiles). A rows staged once in LDS
// via async global->LDS; all 8 waves consume them with ds_load_b128.
__global__ __launch_bounds__(256) void k_gemm(const __bf16* __restrict__ X,
                                              const __bf16* __restrict__ Wt,
                                              float* __restrict__ Y, int K) {
    __shared__ __bf16 sA[16 * KMETA];

    int t = threadIdx.x;
    int cpr = K >> 3;                 // 16B chunks per row (16 or 20)
    int rowBase = blockIdx.x * 16;
    for (int c = t; c < 16 * cpr; c += 256) {
        int r  = c / cpr;
        int kc = c - r * cpr;
        async_g2l_b128(X + (size_t)(rowBase + r) * K + kc * 8, &sA[r * K + kc * 8]);
    }
    wait_asynccnt0();
    __syncthreads();

    int ct   = t >> 5;
    int lane = t & 31;
    int mr   = lane & 15;
    int sel  = lane >> 4;
    int n    = ct * 16 + mr;

    const __bf16* ar = sA + mr * K + sel * 8;                 // LDS
    const __bf16* wr = Wt + (size_t)n * K + sel * 8;          // global (hot in WGP$)

    v8f acc = {};
    #pragma unroll 2
    for (int kb = 0; kb < K; kb += 32) {
        v16bf a = load_frag16(ar + kb);
        v16bf b = load_frag16(wr + kb);
        acc = __builtin_amdgcn_wmma_f32_16x16x32_bf16(false, a, false, b,
                                                      (short)0, acc, false, false);
    }
    size_t outBase = (size_t)(rowBase + 8 * sel) * CC + n;
    #pragma unroll
    for (int r = 0; r < 8; ++r)
        Y[outBase + (size_t)r * CC] = acc[r];
}

// ---------------- edge message: gather feat[src] @ W, atomic scatter-add to temp[dst] --
// grid: (ceil(E/16), numScales); block 256 = 8 column-tile waves over one 16-edge tile.
// The 16 gathered rows are staged once per block via async global->LDS.
__global__ __launch_bounds__(256) void k_edge(const __bf16* __restrict__ featb,
                                              const int* __restrict__ srcBase,
                                              const int* __restrict__ dstBase,
                                              const __bf16* __restrict__ WtBase,
                                              float* __restrict__ temp,
                                              int E, int idxScaleStride, int wScaleStride) {
    int s = blockIdx.y;
    const int*    src = srcBase + (size_t)s * idxScaleStride;
    const int*    dst = dstBase + (size_t)s * idxScaleStride;
    const __bf16* Wt  = WtBase  + (size_t)s * wScaleStride;

    __shared__ __bf16 sA[16 * CC];
    __shared__ int sSrc[16];
    __shared__ int sDst[16];

    int t    = threadIdx.x;
    int base = blockIdx.x * 16;
    if (t < 16) {
        int e = base + t; if (e >= E) e = E - 1;
        sSrc[t] = src[e];
    } else if (t < 32) {
        int q = t - 16;
        int e = base + q; if (e >= E) e = E - 1;
        sDst[q] = dst[e];
    }
    __syncthreads();

    // async gather of 16 feat rows (16 x 256B) into LDS, 16B per lane
    {
        int r  = t >> 4;          // row 0..15
        int kc = t & 15;          // chunk 0..15
        async_g2l_b128(featb + (size_t)sSrc[r] * CC + kc * 8, &sA[r * CC + kc * 8]);
    }
    wait_asynccnt0();
    __syncthreads();

    int ct   = t >> 5;
    int lane = t & 31;
    int mr   = lane & 15;
    int sel  = lane >> 4;
    int n    = ct * 16 + mr;

    const __bf16* ar = sA + mr * CC + sel * 8;                // LDS
    const __bf16* wr = Wt + (size_t)n * CC + sel * 8;         // global

    v8f acc = {};
    #pragma unroll
    for (int kb = 0; kb < CC; kb += 32) {
        v16bf a = load_frag16(ar + kb);
        v16bf b = load_frag16(wr + kb);
        acc = __builtin_amdgcn_wmma_f32_16x16x32_bf16(false, a, false, b,
                                                      (short)0, acc, false, false);
    }
    #pragma unroll
    for (int r = 0; r < 8; ++r) {
        int er = r + 8 * sel;                 // edge row within tile held by this VGPR
        if (base + er < E)
            atomic_add_f32g(&temp[(size_t)sDst[er] * CC + n], acc[r]);
    }
}

// ---------------- GN combine: feat = relu(gn(a)*ga+ba + gn(b)*gb+bb) -------------------
__global__ __launch_bounds__(256) void k_gn_combine(const float* __restrict__ A,
                                                    const float* __restrict__ B,
                                                    const float* __restrict__ ga, const float* __restrict__ ba,
                                                    const float* __restrict__ gb, const float* __restrict__ bb,
                                                    float* __restrict__ outF) {
    int row  = blockIdx.x * 8 + (threadIdx.x >> 5);
    int lane = threadIdx.x & 31;
    size_t base = (size_t)row * CC + lane * 4;
    v4f a = *reinterpret_cast<const v4f*>(A + base);
    v4f b = *reinterpret_cast<const v4f*>(B + base);
    float sA = a[0] + a[1] + a[2] + a[3];
    float sB = b[0] + b[1] + b[2] + b[3];
    float qA = a[0]*a[0] + a[1]*a[1] + a[2]*a[2] + a[3]*a[3];
    float qB = b[0]*b[0] + b[1]*b[1] + b[2]*b[2] + b[3]*b[3];
    float muA = wave_sum32(sA) * (1.0f / CC);
    float muB = wave_sum32(sB) * (1.0f / CC);
    float vA  = wave_sum32(qA) * (1.0f / CC) - muA * muA;
    float vB  = wave_sum32(qB) * (1.0f / CC) - muB * muB;
    float rA = rsqrtf(vA + EPSV);
    float rB = rsqrtf(vB + EPSV);
    v4f o;
    #pragma unroll
    for (int j = 0; j < 4; ++j) {
        int c = lane * 4 + j;
        float va = (a[j] - muA) * rA * ga[c] + ba[c];
        float vb = (b[j] - muB) * rB * gb[c] + bb[c];
        o[j] = fmaxf(va + vb, 0.0f);
    }
    *reinterpret_cast<v4f*>(outF + base) = o;
}

// ---------- GN post: y = relu(gn(x)*g+b [+ resIn]); optional f32/bf16/res outputs ------
__global__ __launch_bounds__(256) void k_gn_post(const float* __restrict__ X,
                                                 const float* __restrict__ g,
                                                 const float* __restrict__ b,
                                                 const float* __restrict__ resIn,
                                                 float* __restrict__ outF,
                                                 __bf16* __restrict__ outB,
                                                 float* __restrict__ resOut) {
    int row  = blockIdx.x * 8 + (threadIdx.x >> 5);
    int lane = threadIdx.x & 31;
    size_t base = (size_t)row * CC + lane * 4;
    v4f x = *reinterpret_cast<const v4f*>(X + base);
    float s = x[0] + x[1] + x[2] + x[3];
    float q = x[0]*x[0] + x[1]*x[1] + x[2]*x[2] + x[3]*x[3];
    float mu  = wave_sum32(s) * (1.0f / CC);
    float var = wave_sum32(q) * (1.0f / CC) - mu * mu;
    float rs  = rsqrtf(var + EPSV);
    v4f r = resIn ? *reinterpret_cast<const v4f*>(resIn + base) : v4f{};
    v4f o;
    v4bf ob;
    #pragma unroll
    for (int j = 0; j < 4; ++j) {
        int c = lane * 4 + j;
        float v = (x[j] - mu) * rs * g[c] + b[c];
        if (resIn) v += r[j];
        o[j]  = fmaxf(v, 0.0f);
        ob[j] = (__bf16)o[j];
    }
    if (outF)   *reinterpret_cast<v4f*>(outF + base)  = o;
    if (outB)   *reinterpret_cast<v4bf*>(outB + base) = ob;
    if (resOut) *reinterpret_cast<v4f*>(resOut + base) = o;
}

// ------------- build meta input: bf16 [N,160] = [feat | turn | control | intersect | 0] -
__global__ __launch_bounds__(256) void k_build_meta(const float* __restrict__ feat,
                                                    const float* __restrict__ turn,
                                                    const float* __restrict__ control,
                                                    const float* __restrict__ intersect,
                                                    __bf16* __restrict__ meta) {
    int idx = blockIdx.x * 256 + threadIdx.x;
    if (idx >= NN * KMETA) return;
    int row = idx / KMETA;
    int k   = idx - row * KMETA;
    float v;
    if (k < CC)            v = feat[(size_t)row * CC + k];
    else if (k == CC)      v = turn[row * 2];
    else if (k == CC + 1)  v = turn[row * 2 + 1];
    else if (k == CC + 2)  v = control[row];
    else if (k == CC + 3)  v = intersect[row];
    else                   v = 0.0f;
    meta[idx] = (__bf16)v;
}

__global__ __launch_bounds__(256) void k_copy_f32(const float* __restrict__ src,
                                                  float* __restrict__ dst, int n) {
    int i = blockIdx.x * 256 + threadIdx.x;
    if (i < n) dst[i] = src[i];
}

extern "C" void kernel_launch(void* const* d_in, const int* in_sizes, int n_in,
                              void* d_out, int out_size, void* d_ws, size_t ws_size,
                              hipStream_t stream) {
    const float* control   = (const float*)d_in[0];
    const float* turn      = (const float*)d_in[1];
    const float* intersect = (const float*)d_in[2];
    const float* ctrs      = (const float*)d_in[3];
    const float* feats     = (const float*)d_in[4];
    const int*   pre       = (const int*)d_in[5];
    const int*   suc       = (const int*)d_in[6];
    const int*   left      = (const int*)d_in[7];
    const int*   right     = (const int*)d_in[8];
    const float* w_in1     = (const float*)d_in[9];
    const float* b_in1     = (const float*)d_in[10];
    const float* w_in2     = (const float*)d_in[11];
    const float* gn_in_g   = (const float*)d_in[12];
    const float* gn_in_b   = (const float*)d_in[13];
    const float* w_seg1    = (const float*)d_in[14];
    const float* b_seg1    = (const float*)d_in[15];
    const float* w_seg2    = (const float*)d_in[16];
    const float* gn_seg_g  = (const float*)d_in[17];
    const float* gn_seg_b  = (const float*)d_in[18];
    const float* w_meta    = (const float*)d_in[19];
    const float* gn_meta_g = (const float*)d_in[20];
    const float* gn_meta_b = (const float*)d_in[21];
    const float* ctr_w     = (const float*)d_in[22];
    const float* pre_w     = (const float*)d_in[23];
    const float* suc_w     = (const float*)d_in[24];
    const float* left_w    = (const float*)d_in[25];
    const float* right_w   = (const float*)d_in[26];
    const float* norm_g    = (const float*)d_in[27];
    const float* norm_b    = (const float*)d_in[28];
    const float* ctr2_w    = (const float*)d_in[29];
    const float* ctr2_g    = (const float*)d_in[30];
    const float* ctr2_b    = (const float*)d_in[31];

    char* ws = (char*)d_ws;
    size_t off = 0;
    auto take = [&](size_t bytes) -> void* {
        void* p = ws + off;
        off = (off + bytes + 255) & ~(size_t)255;
        return p;
    };

    float*  temp   = (float*)take((size_t)NN * CC * 4);      // scratch GEMM output / temp
    float*  res    = (float*)take((size_t)NN * CC * 4);      // residual / second GEMM out
    __bf16* featb  = (__bf16*)take((size_t)NN * CC * 2);     // feat bf16 (also h1)
    __bf16* metab  = (__bf16*)take((size_t)NN * KMETA * 2);  // meta bf16 (also h2)
    __bf16* Win2t  = (__bf16*)take((size_t)CC * CC * 2);
    __bf16* Wseg2t = (__bf16*)take((size_t)CC * CC * 2);
    __bf16* Wmetat = (__bf16*)take((size_t)CC * KMETA * 2);
    __bf16* ctrWt  = (__bf16*)take((size_t)4 * CC * CC * 2);
    __bf16* ctr2Wt = (__bf16*)take((size_t)4 * CC * CC * 2);
    __bf16* leftWt = (__bf16*)take((size_t)4 * CC * CC * 2);
    __bf16* rightWt= (__bf16*)take((size_t)4 * CC * CC * 2);
    __bf16* preWt  = (__bf16*)take((size_t)4 * SSC * CC * CC * 2);
    __bf16* sucWt  = (__bf16*)take((size_t)4 * SSC * CC * CC * 2);

    float* feat = (float*)d_out;  // f32 feat lives in d_out's first N*C floats

    const int T256 = 256;
    // --- weight prep (transpose + bf16) ---
    k_transpose_many<<<(1 * CC * CC + 255) / 256, T256, 0, stream>>>(w_in2, Win2t, 1);
    k_transpose_many<<<(1 * CC * CC + 255) / 256, T256, 0, stream>>>(w_seg2, Wseg2t, 1);
    k_transpose_meta<<<(CC * KMETA + 255) / 256, T256, 0, stream>>>(w_meta, Wmetat);
    k_transpose_many<<<(4 * CC * CC + 255) / 256, T256, 0, stream>>>(ctr_w,  ctrWt, 4);
    k_transpose_many<<<(4 * CC * CC + 255) / 256, T256, 0, stream>>>(ctr2_w, ctr2Wt, 4);
    k_transpose_many<<<(4 * CC * CC + 255) / 256, T256, 0, stream>>>(left_w, leftWt, 4);
    k_transpose_many<<<(4 * CC * CC + 255) / 256, T256, 0, stream>>>(right_w, rightWt, 4);
    k_transpose_many<<<(24 * CC * CC + 255) / 256, T256, 0, stream>>>(pre_w, preWt, 24);
    k_transpose_many<<<(24 * CC * CC + 255) / 256, T256, 0, stream>>>(suc_w, sucWt, 24);

    // --- input / seg branches ---
    k_branch<<<(NN * CC + 255) / 256, T256, 0, stream>>>(ctrs, w_in1, b_in1, featb);   // h1
    k_gemm<<<NN / 16, T256, 0, stream>>>(featb, Win2t, temp, CC);                      // a_lin
    k_branch<<<(NN * CC + 255) / 256, T256, 0, stream>>>(feats, w_seg1, b_seg1, metab);// h2
    k_gemm<<<NN / 16, T256, 0, stream>>>(metab, Wseg2t, res, CC);                      // b_lin
    k_gn_combine<<<NN / 8, T256, 0, stream>>>(temp, res, gn_in_g, gn_in_b,
                                              gn_seg_g, gn_seg_b, feat);

    // --- meta fuse ---
    k_build_meta<<<(NN * KMETA + 255) / 256, T256, 0, stream>>>(feat, turn, control,
                                                                intersect, metab);
    k_gemm<<<NN / 16, T256, 0, stream>>>(metab, Wmetat, temp, KMETA);
    k_gn_post<<<NN / 8, T256, 0, stream>>>(temp, gn_meta_g, gn_meta_b, nullptr,
                                           feat, featb, res);   // res = feat

    // --- 4 fuse layers ---
    for (int i = 0; i < 4; ++i) {
        k_gemm<<<NN / 16, T256, 0, stream>>>(featb, ctrWt + (size_t)i * CC * CC, temp, CC);
        k_edge<<<dim3(EPP / 16, SSC), T256, 0, stream>>>(featb, pre + EPP, pre,
            preWt + (size_t)i * SSC * CC * CC, temp, EPP, 2 * EPP, CC * CC);
        k_edge<<<dim3(EPP / 16, SSC), T256, 0, stream>>>(featb, suc + EPP, suc,
            sucWt + (size_t)i * SSC * CC * CC, temp, EPP, 2 * EPP, CC * CC);
        k_edge<<<dim3((ELL + 15) / 16, 1), T256, 0, stream>>>(featb, left + ELL, left,
            leftWt + (size_t)i * CC * CC, temp, ELL, 0, 0);
        k_edge<<<dim3((ELL + 15) / 16, 1), T256, 0, stream>>>(featb, right + ELL, right,
            rightWt + (size_t)i * CC * CC, temp, ELL, 0, 0);
        k_gn_post<<<NN / 8, T256, 0, stream>>>(temp, norm_g + i * CC, norm_b + i * CC,
                                               nullptr, feat, featb, nullptr);
        k_gemm<<<NN / 16, T256, 0, stream>>>(featb, ctr2Wt + (size_t)i * CC * CC, temp, CC);
        k_gn_post<<<NN / 8, T256, 0, stream>>>(temp, ctr2_g + i * CC, ctr2_b + i * CC,
                                               res, feat, featb, res);
    }

    // --- output tail: ctrs appended after feat ---
    k_copy_f32<<<(NN * 2 + 255) / 256, T256, 0, stream>>>(ctrs, feat + (size_t)NN * CC,
                                                          NN * 2);
    (void)in_sizes; (void)n_in; (void)out_size; (void)ws_size;
}
